// MeMCeNNLayer_79602923864200
// MI455X (gfx1250) — matrix operand order repaired and checked by reading, not compiled, not run on hardware
//
#include <hip/hip_runtime.h>

#define HH 160
#define WW 160
#define NPIX (HH*WW)                       // 25600
#define NELEM ((size_t)8*32*NPIX)          // 6,553,600 per tensor

// padded NHWC f16 activation images: 162 x 162 with 1-pixel zero halo
#define HP 162
#define WP 162
#define PPIX (HP*WP)                       // 26244
#define HALVES_PER_BUF ((size_t)8*PPIX*32) // 6,718,464
#define VEC8_PER_BUF   (HALVES_PER_BUF/8)  // 839,808

#define TIME_STEP 0.01f
#define Bb_C      0.0001f
#define SIGMA_ON  0.45f
#define Y_ON      0.06f
#define SIGMA_P   4e-05f
#define Aa_C      1e-10f
#define SIGMA_OFF 0.013f
#define Y_OFF     1.0f
#define BETA      500.0f
#define GM        0.025f
#define A_MEM     7.2e-06f
#define B_MEM     4.7f
#define ALPHA     0.01f

typedef _Float16 v16h __attribute__((ext_vector_type(16)));
typedef _Float16 h8   __attribute__((ext_vector_type(8)));
typedef float    v8f  __attribute__((ext_vector_type(8)));
typedef float    f4   __attribute__((ext_vector_type(4)));

__device__ __forceinline__ float nonlinf(float t) {
    float y = fminf(t, 1.0f + ALPHA * (t - 1.0f));
    return fmaxf(y, -1.0f + ALPHA * (y + 1.0f));
}
__device__ __forceinline__ float stepf_pos(float t) {
    return t > 0.0f ? 1.0f : (t < 0.0f ? 0.0f : 0.5f);
}
__device__ __forceinline__ float memristor_update(float xc, float vm, float coup) {
    float mem = GM * xc + A_MEM * expf(B_MEM * sqrtf(fabsf(vm))) * (1.0f - xc);
    float im  = mem * vm;
    float p   = im * vm;
    float roff = Y_OFF / xc;
    float ron  = xc / Y_ON;
    float term_off = Aa_C * sinhf(vm / SIGMA_OFF) * expf(-roff * roff) *
                     expf(1.0f / (1.0f + BETA * p)) * stepf_pos(-vm);
    float term_on  = Bb_C * sinhf(vm / SIGMA_ON) * expf(-ron * ron) *
                     expf(p / SIGMA_P) * stepf_pos(vm);
    return xc + TIME_STEP * (term_off + term_on + coup);
}

// ---------------------------------------------------------------------------
// Zero both padded activation buffers (establishes the zero halo; interior is
// overwritten by act_prep every iteration, halo is never touched again).
// ---------------------------------------------------------------------------
__global__ __launch_bounds__(256) void zero_act_kernel(
    _Float16* __restrict__ ibuf, _Float16* __restrict__ i1buf)
{
    size_t i = (size_t)blockIdx.x * 256 + threadIdx.x;
    if (i >= VEC8_PER_BUF) return;
    h8 z = {};
    *(h8*)(ibuf  + i * 8) = z;
    *(h8*)(i1buf + i * 8) = z;
}

// ---------------------------------------------------------------------------
// Pack WA/WB (fp32 [32oc][32ic][3][3]) -> f16 WMMA B-operand layout:
// pw[tap][tile][lane][e], e=0..15:  K(=ic) = (e&7) + (e>=8 ? 16 : 0) + (lane>=16 ? 8 : 0),
// N(=oc) = tile*16 + (lane & 15).   t < 2*9216.
// ---------------------------------------------------------------------------
__global__ __launch_bounds__(256) void pack_weights_kernel(
    const float* __restrict__ WA, const float* __restrict__ WB,
    _Float16* __restrict__ pwA, _Float16* __restrict__ pwB)
{
    int t = blockIdx.x * 256 + threadIdx.x;       // 0 .. 18431
    int which = t / 9216;                          // 0 -> WB, 1 -> WA
    int rem   = t % 9216;
    int e    = rem & 15;
    int lane = (rem >> 4) & 31;
    int tile = (rem >> 9) & 1;
    int tap  = rem >> 10;                          // 0..8
    int oc = tile * 16 + (lane & 15);
    int ic = (e & 7) + ((e >> 3) << 4) + ((lane >= 16) ? 8 : 0);
    int r = tap / 3, s = tap % 3;
    const float* W = which ? WA : WB;
    _Float16* out  = which ? pwA : pwB;
    out[rem] = (_Float16)W[((oc * 32 + ic) * 3 + r) * 3 + s];
}

// ---------------------------------------------------------------------------
// Input convs (3 in-channels, K=27): x' = conv(x,Wx)+bx -> d_out (x half),
// conv_v = conv(v,Wv)+bv -> workspace.  One thread per output element.
// ---------------------------------------------------------------------------
__global__ __launch_bounds__(256) void input_conv_kernel(
    const float* __restrict__ xin, const float* __restrict__ vin,
    const float* __restrict__ Wx, const float* __restrict__ bx,
    const float* __restrict__ Wv, const float* __restrict__ bv,
    float* __restrict__ xout, float* __restrict__ convv)
{
    int idx = blockIdx.x * 256 + threadIdx.x;     // < 6,553,600
    int q  = idx % NPIX;
    int t  = idx / NPIX;
    int oc = t & 31;
    int b  = t >> 5;
    int y = q / WW, x = q % WW;
    float ax = bx[oc], av = bv[oc];
    #pragma unroll
    for (int ic = 0; ic < 3; ++ic) {
        const float* px = xin + (size_t)(b * 3 + ic) * NPIX;
        const float* pv = vin + (size_t)(b * 3 + ic) * NPIX;
        const float* wxp = Wx + (oc * 3 + ic) * 9;
        const float* wvp = Wv + (oc * 3 + ic) * 9;
        #pragma unroll
        for (int r = 0; r < 3; ++r) {
            int sy = y + r - 1;
            if ((unsigned)sy >= (unsigned)HH) continue;
            #pragma unroll
            for (int s = 0; s < 3; ++s) {
                int sx = x + s - 1;
                if ((unsigned)sx >= (unsigned)WW) continue;
                float wa = wxp[r * 3 + s], wb = wvp[r * 3 + s];
                ax += px[sy * WW + sx] * wa;
                av += pv[sy * WW + sx] * wb;
            }
        }
    }
    xout[idx]  = ax;
    convv[idx] = av;
}

// ---------------------------------------------------------------------------
// Per iteration: i = nonlin(conv_v + t*0.01), i1 = nonlin(i); write f16 into
// the HALOED NHWC layout via LDS transpose.  Block = 256 threads handling
// 64 pixels x 32 channels of one image (interior pixels only).
// ---------------------------------------------------------------------------
__global__ __launch_bounds__(256) void act_prep_kernel(
    const float* __restrict__ convv, _Float16* __restrict__ ibuf,
    _Float16* __restrict__ i1buf, float vofs)
{
    __shared__ _Float16 li[64 * 40];
    __shared__ _Float16 li1[64 * 40];
    int b  = blockIdx.x / 400;
    int q0 = (blockIdx.x % 400) * 64;
    int t  = threadIdx.x;
    #pragma unroll
    for (int k = 0; k < 8; ++k) {
        int e  = t + k * 256;          // < 2048
        int ic = e >> 6;
        int pq = e & 63;
        float vt = convv[(size_t)(b * 32 + ic) * NPIX + q0 + pq] + vofs;
        float i0 = nonlinf(vt);
        float i1 = nonlinf(i0);
        li[pq * 40 + ic]  = (_Float16)i0;
        li1[pq * 40 + ic] = (_Float16)i1;
    }
    __syncthreads();
    int pq  = t >> 2;
    int ic0 = (t & 3) * 8;
    h8 o0, o1;
    #pragma unroll
    for (int j = 0; j < 8; ++j) {
        o0[j] = li[pq * 40 + ic0 + j];
        o1[j] = li1[pq * 40 + ic0 + j];
    }
    int q = q0 + pq;
    int y = q / WW, x = q % WW;
    size_t ob = (((size_t)b * HP + (y + 1)) * WP + (x + 1)) * 32 + ic0; // 16B aligned
    *(h8*)(ibuf  + ob) = o0;
    *(h8*)(i1buf + ob) = o1;
}

// ---------------------------------------------------------------------------
// Implicit-GEMM coupling conv via WMMA + memristor x-update.
// One wave per 16-pixel row-tile (all 32 oc). 9 taps x 4 WMMA = 36
// v_wmma_f32_16x16x32_f16 per wave.  Halo makes every A-load unconditional:
// one base address per lane + compile-time immediate offsets per tap.
// Grid = 1600 blocks x 256 threads = 12800 waves = 8*160*10 tiles exactly.
// ---------------------------------------------------------------------------
__global__ __launch_bounds__(256) void conv_update_kernel(
    const _Float16* __restrict__ ibuf, const _Float16* __restrict__ i1buf,
    const _Float16* __restrict__ pwB, const _Float16* __restrict__ pwA,
    const float* __restrict__ bA, const float* __restrict__ bB,
    const float* __restrict__ Zb,
    float* __restrict__ xbuf, const float* __restrict__ convv,
    float* __restrict__ vout, float vofs, int last)
{
    int wid  = (blockIdx.x * 256 + threadIdx.x) >> 5;   // 0..12799
    int lane = threadIdx.x & 31;
    int b   = wid / 1600;
    int rem = wid % 1600;
    int y   = rem / 10;
    int x0  = (rem % 10) * 16;
    int ln  = lane & 15;
    int hi  = lane >> 4;

    v8f acc0, acc1;
    {
        float c0 = bA[ln] + bB[ln] + Zb[ln];
        float c1 = bA[ln + 16] + bB[ln + 16] + Zb[ln + 16];
        #pragma unroll
        for (int g = 0; g < 8; ++g) { acc0[g] = c0; acc1[g] = c1; }
    }

    const v16h* wB = (const v16h*)pwB;
    const v16h* wA = (const v16h*)pwA;

    // Per-lane base: padded pixel (y, x0+ln) = tap (r=0,s=0) source.
    // A layout: lane<16 holds K 0-7 / 16-23, lane>=16 holds K 8-15 / 24-31.
    size_t pb0 = (((size_t)b * HP + y) * WP + (x0 + ln)) * 32 + hi * 8;
    const _Float16* pi  = ibuf  + pb0;
    const _Float16* pi1 = i1buf + pb0;

    #pragma unroll
    for (int tap = 0; tap < 9; ++tap) {
        const int r = tap / 3, s = tap % 3;
        const int ofs = (r * WP + s) * 32;          // halves; <= 10432
        h8 a_lo = *(const h8*)(pi  + ofs);
        h8 a_hi = *(const h8*)(pi  + ofs + 16);
        h8 c_lo = *(const h8*)(pi1 + ofs);
        h8 c_hi = *(const h8*)(pi1 + ofs + 16);
        v16h Ai  = __builtin_shufflevector(a_lo, a_hi, 0,1,2,3,4,5,6,7,8,9,10,11,12,13,14,15);
        v16h Ai1 = __builtin_shufflevector(c_lo, c_hi, 0,1,2,3,4,5,6,7,8,9,10,11,12,13,14,15);
        v16h B0 = wB[(tap * 2 + 0) * 32 + lane];
        v16h B1 = wB[(tap * 2 + 1) * 32 + lane];
        v16h C0 = wA[(tap * 2 + 0) * 32 + lane];
        v16h C1 = wA[(tap * 2 + 1) * 32 + lane];
        acc0 = __builtin_amdgcn_wmma_f32_16x16x32_f16(false, Ai,  false, B0, (short)0, acc0, false, false);
        acc1 = __builtin_amdgcn_wmma_f32_16x16x32_f16(false, Ai,  false, B1, (short)0, acc1, false, false);
        acc0 = __builtin_amdgcn_wmma_f32_16x16x32_f16(false, Ai1, false, C0, (short)0, acc0, false, false);
        acc1 = __builtin_amdgcn_wmma_f32_16x16x32_f16(false, Ai1, false, C1, (short)0, acc1, false, false);
    }

    // C/D layout: VGPR g, lane -> pixel M = g + hi*8, oc = tile*16 + ln.
    int px0 = x0 + hi * 8;
    #pragma unroll
    for (int tile = 0; tile < 2; ++tile) {
        int oc = tile * 16 + ln;
        v8f acc = tile ? acc1 : acc0;
        size_t base = ((size_t)(b * 32 + oc) * HH + y) * WW + px0;  // 16B aligned
        f4 xv0 = *(const f4*)(xbuf + base);
        f4 xv1 = *(const f4*)(xbuf + base + 4);
        f4 cv0 = *(const f4*)(convv + base);
        f4 cv1 = *(const f4*)(convv + base + 4);
        f4 nx0, nx1, nv0, nv1;
        #pragma unroll
        for (int g = 0; g < 8; ++g) {
            float xc = (g < 4) ? xv0[g] : xv1[g - 4];
            float vm = ((g < 4) ? cv0[g] : cv1[g - 4]) + vofs;
            float nx = memristor_update(xc, vm, acc[g]);
            if (g < 4) { nx0[g] = nx; nv0[g] = vm + TIME_STEP; }
            else       { nx1[g - 4] = nx; nv1[g - 4] = vm + TIME_STEP; }
        }
        *(f4*)(xbuf + base)     = nx0;
        *(f4*)(xbuf + base + 4) = nx1;
        if (last) {
            *(f4*)(vout + base)     = nv0;
            *(f4*)(vout + base + 4) = nv1;
        }
    }
}

extern "C" void kernel_launch(void* const* d_in, const int* in_sizes, int n_in,
                              void* d_out, int out_size, void* d_ws, size_t ws_size,
                              hipStream_t stream)
{
    (void)in_sizes; (void)n_in; (void)out_size; (void)ws_size;
    const float* x  = (const float*)d_in[0];
    const float* v  = (const float*)d_in[1];
    const float* Wx = (const float*)d_in[2];
    const float* bx = (const float*)d_in[3];
    const float* Wv = (const float*)d_in[4];
    const float* bv = (const float*)d_in[5];
    const float* WA = (const float*)d_in[6];
    const float* bA = (const float*)d_in[7];
    const float* WB = (const float*)d_in[8];
    const float* bB = (const float*)d_in[9];
    const float* Zb = (const float*)d_in[10];

    float* xout = (float*)d_out;           // x occupies first half of d_out
    float* vout = xout + NELEM;            // v second half

    char* ws = (char*)d_ws;
    float*    convv = (float*)ws;                              // 26,214,400 B
    _Float16* ibuf  = (_Float16*)(ws + 26214400);              // 13,436,928 B (haloed)
    _Float16* i1buf = (_Float16*)(ws + 39651328);              // 13,436,928 B (haloed)
    _Float16* pwB   = (_Float16*)(ws + 53088256);              //     18,432 B
    _Float16* pwA   = (_Float16*)(ws + 53088256 + 32768);      //     18,432 B

    zero_act_kernel<<<(int)((VEC8_PER_BUF + 255) / 256), 256, 0, stream>>>(ibuf, i1buf);
    pack_weights_kernel<<<72, 256, 0, stream>>>(WA, WB, pwA, pwB);
    input_conv_kernel<<<25600, 256, 0, stream>>>(x, v, Wx, bx, Wv, bv, xout, convv);

    for (int t = 0; t < 10; ++t) {
        float vofs = (float)t * TIME_STEP;
        act_prep_kernel<<<3200, 256, 0, stream>>>(convv, ibuf, i1buf, vofs);
        conv_update_kernel<<<1600, 256, 0, stream>>>(ibuf, i1buf, pwB, pwA,
                                                     bA, bB, Zb,
                                                     xout, convv, vout,
                                                     vofs, (t == 9) ? 1 : 0);
    }
}
